// LstmTl_18631568130324
// MI455X (gfx1250) — matrix operand order repaired and checked by reading, not compiled, not run on hardware
//
#include <hip/hip_runtime.h>
#include <hip/hip_bf16.h>

// ---------------------------------------------------------------------------
// LSTM (B=64, T=2048, D=H=256) for gfx1250 / MI455X.
//
// Phase 1 (pack_weights): convert [Wx;Wh] (f32, 512x1024) into bf16 WMMA
//   B-fragment order in d_ws (1 MB, L2-resident), and b = bx+bh (f32).
// Phase 2 (lstm_persistent): 4 persistent WGs x 512 threads. WG g owns batch
//   rows [16g,16g+16). Per step: gates = [x_t|h] @ Wpack + b via
//   v_wmma_f32_16x16x32_bf16 (K=512 -> 16 chunks), lane-local gate combine,
//   h re-staged to LDS in A-fragment order for the next step.
// ---------------------------------------------------------------------------

typedef __attribute__((ext_vector_type(16))) __bf16 v16bf;
typedef __attribute__((ext_vector_type(8)))  float  v8f;

#define T_STEPS 2048
#define D_IN    256
#define H_DIM   256
#define KTOT    512      // D_IN + H_DIM
#define NKCH    16       // KTOT / 32
#define NTILES  64       // 4H / 16

__device__ __forceinline__ float fast_sigmoid(float x) {
  // 1 / (1 + 2^(-x*log2(e)))
  return __builtin_amdgcn_rcpf(1.0f +
         __builtin_amdgcn_exp2f(x * -1.44269504088896340736f));
}
__device__ __forceinline__ float fast_tanh(float x) {
  // tanh(x) = 2*sigmoid(2x) - 1
  float s = __builtin_amdgcn_rcpf(1.0f +
            __builtin_amdgcn_exp2f(x * -2.88539008177792681472f));
  return 2.0f * s - 1.0f;
}

// --------------------------- Phase 1: weight packing ------------------------
// wpack element offset = ((nt*16 + c)*32 + lane)*16 + (r*2 + e)
// B-fragment layout (32x16, K-major per lane): lane&15 = N column,
// element idx holds kk = 16*(idx>>3) + 8*(lane>>4) + (idx&7), VGPR r packs 2 kk.
__global__ void pack_weights(const float* __restrict__ Wx,
                             const float* __restrict__ Wh,
                             const float* __restrict__ bx,
                             const float* __restrict__ bh,
                             __bf16* __restrict__ wpack,
                             float* __restrict__ bcomb) {
  int gid = blockIdx.x * blockDim.x + threadIdx.x;   // 0 .. 524287
  int idx  = gid & 15;          // r*2 + e
  int lane = (gid >> 4) & 31;
  int c    = (gid >> 9) & 15;   // K chunk
  int nt   = gid >> 13;         // 0..63 gate N-tile
  int n  = nt * 16 + (lane & 15);
  int kk = ((idx >> 3) << 4) + ((lane >> 4) << 3) + (idx & 7);   // 0..31 within chunk
  int k  = c * 32 + kk;                // 0..511
  float v = (k < D_IN) ? Wx[(size_t)k * 1024 + n]
                       : Wh[(size_t)(k - D_IN) * 1024 + n];
  wpack[gid] = (__bf16)v;
  if (gid < 1024) bcomb[gid] = bx[gid] + bh[gid];
}

// --------------------------- Phase 2: persistent LSTM -----------------------
__global__ __launch_bounds__(512)
void lstm_persistent(const float* __restrict__ x,
                     const __bf16* __restrict__ wpack,
                     const float* __restrict__ bias,
                     float* __restrict__ out) {
  // A-fragment staging: 16 K-chunks x (32 lanes x 16 bf16) = 16 KB
  __shared__ __bf16 afrag[NKCH * 512];

  const int tid  = threadIdx.x;
  const int wave = tid >> 5;          // 0..15 : hidden tile
  const int lane = tid & 31;
  const int g    = blockIdx.x;        // 0..3 : batch slice
  const int bRow0 = g * 16;

  // zero the h fragments (chunks 8..15)
  #pragma unroll
  for (int i = 0; i < 8; ++i)
    afrag[8 * 512 + tid * 8 + i] = (__bf16)0.0f;
  __syncthreads();

  const int n    = lane & 15;         // N within tile / C-D column
  const int jcol = wave * 16 + n;     // hidden column 0..255
  const float bF = bias[jcol];
  const float bI = bias[256 + jcol];
  const float bO = bias[512 + jcol];
  const float bC = bias[768 + jcol];

  v8f c_state = {};
  v8f h_state = {};

  // x staging: thread handles row sm, 8 consecutive d starting at sd0.
  const int sm  = tid >> 5;           // 0..15
  const int sd0 = (lane) * 8;         // 0..248

  // h staging constants (A-fragment mapping for k = 256 + jcol):
  const int kk_h   = ((wave & 1) << 4) | n;                  // k & 31
  const int ch_h   = 8 + (wave >> 1);                        // chunk 8..15
  const int ie_h   = ((kk_h >> 4) << 3) + (kk_h & 7);        // (r*2+e) index
  const int lhi_h  = (kk_h >> 3) & 1;

  for (int t = 0; t < T_STEPS; ++t) {
    // ---- stage x_t (chunks 0..7) in A-fragment order ----
    {
      const float* xp = x + ((size_t)(bRow0 + sm) * T_STEPS + t) * D_IN + sd0;
      float4 x0 = *(const float4*)xp;
      float4 x1 = *(const float4*)(xp + 4);
      if (t + 1 < T_STEPS)
        __builtin_prefetch(xp + D_IN, 0, 0);   // next step's x row -> L2
      float xv[8] = {x0.x, x0.y, x0.z, x0.w, x1.x, x1.y, x1.z, x1.w};
      // A layout: lane' = m + 16*((kk>>3)&1); idx = (kk>>4)*8 + (kk&7)
      // for 8-aligned runs: lane' and (kk>>4) fixed -> 16 contiguous bytes
      const int kk0  = sd0 & 31;
      const int cch  = sd0 >> 5;
      const int lp   = sm + (((kk0 >> 3) & 1) << 4);
      const int ibas = ((kk0 >> 4) << 3);
      __bf16* dst = &afrag[cch * 512 + lp * 16 + ibas];
      #pragma unroll
      for (int j = 0; j < 8; ++j) dst[j] = (__bf16)xv[j];
    }
    __syncthreads();   // x staged; h (chunks 8..15) staged last iteration

    // ---- gates = [x_t | h] @ Wpack : K = 512 = 16 chunks of 32 ----
    v8f accF = {}, accI = {}, accO = {}, accC = {};
    #pragma unroll 4
    for (int cc = 0; cc < NKCH; ++cc) {
      v16bf a = *(const v16bf*)(&afrag[cc * 512 + lane * 16]);
      const __bf16* wp = wpack + (size_t)cc * 512 + lane * 16;
      v16bf bf0 = *(const v16bf*)(wp + (size_t)(wave     ) * 16 * 512);
      v16bf bf1 = *(const v16bf*)(wp + (size_t)(wave + 16) * 16 * 512);
      v16bf bf2 = *(const v16bf*)(wp + (size_t)(wave + 32) * 16 * 512);
      v16bf bf3 = *(const v16bf*)(wp + (size_t)(wave + 48) * 16 * 512);
      accF = __builtin_amdgcn_wmma_f32_16x16x32_bf16(false, a, false, bf0,
                                                     (short)0, accF, false, false);
      accI = __builtin_amdgcn_wmma_f32_16x16x32_bf16(false, a, false, bf1,
                                                     (short)0, accI, false, false);
      accO = __builtin_amdgcn_wmma_f32_16x16x32_bf16(false, a, false, bf2,
                                                     (short)0, accO, false, false);
      accC = __builtin_amdgcn_wmma_f32_16x16x32_bf16(false, a, false, bf3,
                                                     (short)0, accC, false, false);
    }

    // ---- lane-local gate combine (C/D layout: m = v + 8*(lane>=16)) ----
    #pragma unroll
    for (int v = 0; v < 8; ++v) {
      float fg = fast_sigmoid(accF[v] + bF);
      float ig = fast_sigmoid(accI[v] + bI);
      float og = fast_sigmoid(accO[v] + bO);
      float cb = fast_tanh(accC[v] + bC);
      float cn = fg * c_state[v] + ig * cb;
      c_state[v] = cn;
      h_state[v] = og * fast_tanh(cn);
    }

    __syncthreads();   // all waves finished reading afrag for step t

    // ---- write h to out; re-stage h (bf16) into chunks 8..15 ----
    #pragma unroll
    for (int v = 0; v < 8; ++v) {
      int m  = v + ((lane >> 4) << 3);
      int lp = m + (lhi_h << 4);
      afrag[ch_h * 512 + lp * 16 + ie_h] = (__bf16)h_state[v];
      out[((size_t)(bRow0 + m) * T_STEPS + t) * H_DIM + jcol] = h_state[v];
    }
  }

  // ---- final states: h_T then c_T appended after outputs ----
  float* hT = out + (size_t)64 * T_STEPS * H_DIM;
  float* cT = hT + 64 * H_DIM;
  #pragma unroll
  for (int v = 0; v < 8; ++v) {
    int m = v + ((lane >> 4) << 3);
    hT[(size_t)(bRow0 + m) * H_DIM + jcol] = h_state[v];
    cT[(size_t)(bRow0 + m) * H_DIM + jcol] = c_state[v];
  }
}

// --------------------------------- launch ----------------------------------
extern "C" void kernel_launch(void* const* d_in, const int* in_sizes, int n_in,
                              void* d_out, int out_size, void* d_ws, size_t ws_size,
                              hipStream_t stream) {
  const float* x  = (const float*)d_in[0];
  const float* Wx = (const float*)d_in[1];
  const float* Wh = (const float*)d_in[2];
  const float* bx = (const float*)d_in[3];
  const float* bh = (const float*)d_in[4];

  __bf16* wpack = (__bf16*)d_ws;                         // 1 MB bf16 fragments
  float*  bcomb = (float*)((char*)d_ws + (size_t)NTILES * NKCH * 512 * sizeof(__bf16));

  // 64 tiles * 16 chunks * 512 elems = 524288 threads
  pack_weights<<<2048, 256, 0, stream>>>(Wx, Wh, bx, bh, wpack, bcomb);
  lstm_persistent<<<4, 512, 0, stream>>>(x, wpack, bcomb, (float*)d_out);
}
